// QuantumLayer1_66211215835678
// MI455X (gfx1250) — matrix operand order repaired and checked by reading, not compiled, not run on hardware
//
#include <hip/hip_runtime.h>

typedef __attribute__((ext_vector_type(16))) _Float16 v16h;
typedef __attribute__((ext_vector_type(8)))  _Float16 v8h;
typedef __attribute__((ext_vector_type(8)))  float    v8f;
typedef __attribute__((ext_vector_type(4)))  float    v4f;

#define NQ  4
#define DIM 16   // 2^NQ

// ---------------------------------------------------------------------------
// Builder: fold the whole 2-layer circuit into a 16x16 complex unitary U.
// Thread j evolves basis state e_j -> column j of U. Writes four f16 matrices
// (row-major [n][k]) into d_ws: Ur_hi, Ur_lo, Ui_hi, Ui_lo (hi/lo f16 split
// of each fp32 entry for error-compensated WMMA).
// ---------------------------------------------------------------------------
__global__ void build_unitary_kernel(const float* __restrict__ w,
                                     _Float16* __restrict__ bmat) {
  int j = threadIdx.x;
  if (j >= DIM) return;
  float re[DIM], im[DIM];
#pragma unroll
  for (int i = 0; i < DIM; ++i) { re[i] = (i == j) ? 1.f : 0.f; im[i] = 0.f; }

#pragma unroll
  for (int layer = 0; layer < 2; ++layer) {
#pragma unroll
    for (int q = 0; q < NQ; ++q) {
      const float th  = w[layer * (NQ * 2) + q * 2 + 0];
      const float phi = w[layer * (NQ * 2) + q * 2 + 1];
      const float ch = cosf(0.5f * th), sh = sinf(0.5f * th);
      const int m = 8 >> q;   // qubit q <-> bit (3-q); wire 0 is MSB
      // RY(th): [[c,-s],[s,c]] on (i0, i0|m) pairs
#pragma unroll
      for (int i = 0; i < DIM; ++i) {
        if (!(i & m)) {
          const int i1 = i | m;
          float r0 = re[i], r1 = re[i1];
          re[i]  = ch * r0 - sh * r1;
          re[i1] = sh * r0 + ch * r1;
          float m0 = im[i], m1 = im[i1];
          im[i]  = ch * m0 - sh * m1;
          im[i1] = sh * m0 + ch * m1;
        }
      }
      // RZ(phi): bit0 -> exp(-i phi/2), bit1 -> exp(+i phi/2)
      const float cp = cosf(0.5f * phi), sp = sinf(0.5f * phi);
#pragma unroll
      for (int i = 0; i < DIM; ++i) {
        const float s = (i & m) ? sp : -sp;
        const float r = re[i], v = im[i];
        re[i] = r * cp - v * s;
        im[i] = r * s + v * cp;
      }
    }
    // CNOT chain: control q, target q+1 -> swap target amps where control=1
#pragma unroll
    for (int q = 0; q < NQ - 1; ++q) {
      const int mc = 8 >> q, mt = 8 >> (q + 1);
#pragma unroll
      for (int i = 0; i < DIM; ++i) {
        if ((i & mc) && !(i & mt)) {
          const int i1 = i | mt;
          float t = re[i]; re[i] = re[i1]; re[i1] = t;
          t = im[i]; im[i] = im[i1]; im[i1] = t;
        }
      }
    }
  }
  // Column j of U -> hi/lo f16 split, row-major [n][k]
#pragma unroll
  for (int n = 0; n < DIM; ++n) {
    const float r = re[n];
    const _Float16 rh = (_Float16)r;
    const _Float16 rl = (_Float16)(r - (float)rh);
    const float c = im[n];
    const _Float16 ih = (_Float16)c;
    const _Float16 il = (_Float16)(c - (float)ih);
    bmat[0 * 256 + n * DIM + j] = rh;   // Ur_hi
    bmat[1 * 256 + n * DIM + j] = rl;   // Ur_lo
    bmat[2 * 256 + n * DIM + j] = ih;   // Ui_hi
    bmat[3 * 256 + n * DIM + j] = il;   // Ui_lo
  }
}

// ---------------------------------------------------------------------------
// Main kernel (transposed GEMM): Y[16 basis, 16 samples] = U * X~ per tile via
// v_wmma_f32_16x16x32_f16. K0..15 = hi(x features), K16..31 = lo(x features);
// two accumulating WMMAs per (re/im) give U*x to ~fp32 accuracy.
// C-layout puts each sample in one lane => basis reduction is an in-register
// 8-point signed WHT + 4 shfl_xor(16); unitarity supplies the normalizer.
// ---------------------------------------------------------------------------
__global__ __launch_bounds__(256) void qlayer_wmma_kernel(
    const float* __restrict__ x, const _Float16* __restrict__ bmat,
    float* __restrict__ out, int ntiles) {
  const int lane = threadIdx.x & 31;
  const int wavesPerBlock = blockDim.x >> 5;
  const int gw = blockIdx.x * wavesPerBlock + (threadIdx.x >> 5);
  const int nw = gridDim.x * wavesPerBlock;
  const int ln   = lane & 15;
  const int half = lane >> 4;

  // A-operands (constant per wave): lane holds rows n=ln; element s is
  // U column s (K = 16*half + s) -> same U columns in both halves (hi/lo split)
  const int aoff = ln * DIM;
  const v16h urh = *(const v16h*)(bmat + 0 * 256 + aoff);
  const v16h url = *(const v16h*)(bmat + 1 * 256 + aoff);
  const v16h uih = *(const v16h*)(bmat + 2 * 256 + aoff);
  const v16h uil = *(const v16h*)(bmat + 3 * 256 + aoff);
  const v16h a_rh = __builtin_shufflevector(urh, urh, 0,1,2,3,4,5,6,7, 8,9,10,11,12,13,14,15);
  const v16h a_rl = __builtin_shufflevector(url, url, 0,1,2,3,4,5,6,7, 8,9,10,11,12,13,14,15);
  const v16h a_ih = __builtin_shufflevector(uih, uih, 0,1,2,3,4,5,6,7, 8,9,10,11,12,13,14,15);
  const v16h a_il = __builtin_shufflevector(uil, uil, 0,1,2,3,4,5,6,7, 8,9,10,11,12,13,14,15);

  for (int t = gw; t < ntiles; t += nw) {
    const int s0 = t << 4;
    // B-operand: column N=ln is sample s0+ln; lanes 0-15 carry hi(x),
    // lanes 16-31 carry lo(x) (K=16..31)
    const float* xp = x + (size_t)(s0 + ln) * DIM;
    const v4f x0 = *(const v4f*)(xp);
    const v4f x1 = *(const v4f*)(xp + 4);
    const v4f x2 = *(const v4f*)(xp + 8);
    const v4f x3 = *(const v4f*)(xp + 12);

    v16h b;
#pragma unroll
    for (int i = 0; i < 4; ++i) {
      float f;
      f = x0[i]; { _Float16 h = (_Float16)f; _Float16 l = (_Float16)(f - (float)h); b[i +  0] = half ? l : h; }
      f = x1[i]; { _Float16 h = (_Float16)f; _Float16 l = (_Float16)(f - (float)h); b[i +  4] = half ? l : h; }
      f = x2[i]; { _Float16 h = (_Float16)f; _Float16 l = (_Float16)(f - (float)h); b[i +  8] = half ? l : h; }
      f = x3[i]; { _Float16 h = (_Float16)f; _Float16 l = (_Float16)(f - (float)h); b[i + 12] = half ? l : h; }
    }

    v8f cre = {}; v8f cim = {};
    cre = __builtin_amdgcn_wmma_f32_16x16x32_f16(false, a_rh, false, b, (short)0, cre, false, false);
    cre = __builtin_amdgcn_wmma_f32_16x16x32_f16(false, a_rl, false, b, (short)0, cre, false, false);
    cim = __builtin_amdgcn_wmma_f32_16x16x32_f16(false, a_ih, false, b, (short)0, cim, false, false);
    cim = __builtin_amdgcn_wmma_f32_16x16x32_f16(false, a_il, false, b, (short)0, cim, false, false);

    // probs for basis M = v + 8*half of this lane's sample
    float p0 = cre[0]*cre[0] + cim[0]*cim[0];
    float p1 = cre[1]*cre[1] + cim[1]*cim[1];
    float p2 = cre[2]*cre[2] + cim[2]*cim[2];
    float p3 = cre[3]*cre[3] + cim[3]*cim[3];
    float p4 = cre[4]*cre[4] + cim[4]*cim[4];
    float p5 = cre[5]*cre[5] + cim[5]*cim[5];
    float p6 = cre[6]*cre[6] + cim[6]*cim[6];
    float p7 = cre[7]*cre[7] + cim[7]*cim[7];

    // in-register 8-point WHT over v (basis bits 2,1,0 -> qubits 1,2,3)
    const float e0 = p0 + p1, o0 = p0 - p1;
    const float e1 = p2 + p3, o1 = p2 - p3;
    const float e2 = p4 + p5, o2 = p4 - p5;
    const float e3 = p6 + p7, o3 = p6 - p7;
    const float A3 = (o0 + o1) + (o2 + o3);          // sign = bit0(M) -> qubit 3
    const float f0 = e0 + e1, g0 = e0 - e1;
    const float f1 = e2 + e3, g1 = e2 - e3;
    const float A2 = g0 + g1;                        // sign = bit1(M) -> qubit 2
    const float S  = f0 + f1;                        // plain sum over 8 basis
    const float A1 = f0 - f1;                        // sign = bit2(M) -> qubit 1

    // combine halves: bit3(M) == half -> qubit 0 is the low/high split
    const float Sp  = __shfl_xor(S,  16, 32);
    const float A1p = __shfl_xor(A1, 16, 32);
    const float A2p = __shfl_xor(A2, 16, 32);
    const float A3p = __shfl_xor(A3, 16, 32);

    if (!half) {
      const float tot = S + Sp;                      // = ||x||^2 (unitarity) > 0
      // fast reciprocal: v_rcp_f32 + 1 Newton step (~0.5 ulp for this range)
      float inv = __builtin_amdgcn_rcpf(tot);
      inv = __builtin_fmaf(__builtin_fmaf(-tot, inv, 1.0f), inv, inv);
      v4f o;
      o[0] = (S  - Sp ) * inv;                       // qubit 0
      o[1] = (A1 + A1p) * inv;                       // qubit 1
      o[2] = (A2 + A2p) * inv;                       // qubit 2
      o[3] = (A3 + A3p) * inv;                       // qubit 3
      *(v4f*)(out + (size_t)(s0 + ln) * 4) = o;      // coalesced b128 store
    }
  }
}

extern "C" void kernel_launch(void* const* d_in, const int* in_sizes, int n_in,
                              void* d_out, int out_size, void* d_ws, size_t ws_size,
                              hipStream_t stream) {
  const float* x = (const float*)d_in[0];     // [B,16] fp32
  const float* w = (const float*)d_in[1];     // [2,4,2] fp32
  float* out = (float*)d_out;                 // [B,4] fp32
  _Float16* bmat = (_Float16*)d_ws;           // 1024 halves = 2 KB

  const int B = in_sizes[0] / DIM;
  const int ntiles = B / 16;

  build_unitary_kernel<<<1, 32, 0, stream>>>(w, bmat);

  const int wavesPerBlock = 8;                // 256 threads = 8 wave32
  int blocks = (ntiles + wavesPerBlock - 1) / wavesPerBlock;
  if (blocks > 4096) blocks = 4096;           // ~32K waves, grid-stride loop
  if (blocks < 1) blocks = 1;
  qlayer_wmma_kernel<<<blocks, 256, 0, stream>>>(x, bmat, out, ntiles);
}
